// SMCN_36197984370917
// MI455X (gfx1250) — matrix operand order repaired and checked by reading, not compiled, not run on hardware
//
#include <hip/hip_runtime.h>
#include <hip/hip_bf16.h>
#include <math.h>

// Reference shapes: T, BS, N, H, OUT = 128, 64, 64, 256, 64
static constexpr int kT   = 128;
static constexpr int kBS  = 64;
static constexpr int kNP  = 64;   // particles
static constexpr int kH   = 256;  // hidden width
static constexpr int kOUT = 64;   // output width

typedef __bf16 bf16_t;
typedef bf16_t v16bf __attribute__((ext_vector_type(16)));
typedef bf16_t v8bf  __attribute__((ext_vector_type(8)));
typedef float  v8f   __attribute__((ext_vector_type(8)));

// ---------------- deterministic counter-based RNG ----------------
__device__ __forceinline__ uint32_t hash_u32(uint32_t x) {
  x ^= x >> 17; x *= 0xed5ad4bbu;
  x ^= x >> 11; x *= 0xac4c1b51u;
  x ^= x >> 15; x *= 0x31848babu;
  x ^= x >> 14; return x;
}
__device__ __forceinline__ float uniform01(uint32_t h) {
  return (h + 1u) * 2.3283063e-10f;  // (0,1]
}
__device__ __forceinline__ float gaussian(uint32_t seed) {
  float u1 = uniform01(hash_u32(seed * 2u + 1u));
  float u2 = uniform01(hash_u32(seed * 2u + 2u));
  float r  = __builtin_amdgcn_sqrtf(-2.0f * __logf(u1));
  return r * __cosf(6.28318530718f * u2);
}
// branchless tanh: 1 - 2/(e^{2x}+1)   (v_exp + v_add + v_rcp + v_fma)
__device__ __forceinline__ float fast_tanh(float x) {
  float e = __expf(2.0f * x);
  return 1.0f - 2.0f * __builtin_amdgcn_rcpf(e + 1.0f);
}

// ---------------- LDS layout ----------------
// One persistent workgroup per batch element. Particle state and activations
// live in LDS as bf16 (they are only consumed as WMMA A-operands); y_hat stays
// f32 for the log-prob + global output. Weights live in global memory as
// pre-converted bf16 (288KB -> permanently L2-resident, shared by all 64 WGs;
// B-fragments are single contiguous 32B loads).
struct SmcLds {
  alignas(32) bf16_t xbuf[kNP * kH];   // 32KB particle state / FFN z
  alignas(32) bf16_t hbuf[kNP * kH];   // 32KB tanh(...)+noise
  alignas(32) float  ybuf[kNP * kOUT]; // 16KB y_hat
  float su[kH];        // u_t row
  float suproj[kH];    // u W_ih^T + b_ih + b_hh
  float syrow[kOUT];
  float sbih[kH];
  float sb1[kH];
  float sb2[kOUT];
  float sisy[kOUT];
  float ssx[kH];
  float slp[kNP];
  float scdf[kNP];
  int   sI[kNP];
  float stot;
};

// ---------------- 64xK GEMM on one workgroup (8 wave32) ----------------
// C[64][NCOLS] = A[64][kH] * Bt^T   (Bt is [NCOLS][kH] row-major = WMMA B
// operand stored N-major so a lane's 16 K-contiguous bf16 are one 32B read).
//
// Fragment layouts used here (contiguous K order, same for A and B):
//   A (16-bit 16x32): lane 0-15 -> M=lane; lanes 16-31 -> M=lane-16, +16 K-off.
//     element e<8  -> K = kb + 16*half + e
//     element e>=8 -> K = kb + 16*half + e     => two 16B LDS runs
//   B (16-bit 32x16): lane -> N=lane&15; element e -> K = kb + 16*half + e
//     => one contiguous 32B run in the N-major weight row.
//
// MODE 0: fast_tanh(acc + uproj[col])  -> bf16 C   (noise added in later pass)
// MODE 1: relu(acc + b1[col])          -> bf16 C
// MODE 2: acc + b2[col]                -> f32 C and global out
template <int MODE, int NCOLS>
__device__ __forceinline__ void gemm_block(const bf16_t* __restrict__ A,
                                           const bf16_t* __restrict__ Bt,
                                           void* __restrict__ Cout, int tid,
                                           const float* __restrict__ bias,
                                           float* __restrict__ gout) {
  const int wave  = tid >> 5;
  const int lane  = tid & 31;
  const int lrow  = lane & 15;
  const int lhalf = lane >> 4;
  constexpr int TILES_N = NCOLS / 16;
  constexpr int NTILES  = 4 * TILES_N;  // 4 M-tiles (64 rows)
#pragma unroll
  for (int tile = wave; tile < NTILES; tile += 8) {
    const int tm = tile / TILES_N;
    const int tn = tile % TILES_N;
    v8f acc = {};
    const bf16_t* arow = A  + (tm * 16 + lrow) * kH + 16 * lhalf;
    const bf16_t* brow = Bt + (tn * 16 + lrow) * kH + 16 * lhalf;
#pragma unroll
    for (int kb = 0; kb < kH; kb += 32) {
      v8bf lo = *(const v8bf*)(arow + kb);        // ds_load_b128
      v8bf hi = *(const v8bf*)(arow + kb + 8);    // ds_load_b128
      v16bf af = __builtin_shufflevector(lo, hi, 0, 1, 2, 3, 4, 5, 6, 7,
                                         8, 9, 10, 11, 12, 13, 14, 15);
      v16bf bfr = *(const v16bf*)(brow + kb);     // 32B global (L2-resident)
      acc = __builtin_amdgcn_wmma_f32_16x16x32_bf16(false, af, false, bfr,
                                                    (short)0, acc, false, false);
    }
    const int col = tn * 16 + lrow;
#pragma unroll
    for (int r = 0; r < 8; ++r) {
      const int row = tm * 16 + 8 * lhalf + r;
      float v = acc[r] + bias[col];
      if (MODE == 0) {
        ((bf16_t*)Cout)[row * NCOLS + col] = (bf16_t)fast_tanh(v);
      } else if (MODE == 1) {
        ((bf16_t*)Cout)[row * NCOLS + col] = (bf16_t)fmaxf(v, 0.0f);
      } else {
        ((float*)Cout)[row * NCOLS + col] = v;
        gout[row * NCOLS + col] = v;
      }
    }
  }
}

// ---------------- persistent particle-filter kernel ----------------
__global__ __launch_bounds__(256) void smcn_kernel(
    const float* __restrict__ u, const float* __restrict__ y,
    const float* __restrict__ W_ih,
    const float* __restrict__ b_ih, const float* __restrict__ b_hh,
    const float* __restrict__ b1, const float* __restrict__ b2,
    const float* __restrict__ sigx, const float* __restrict__ sigy,
    const bf16_t* __restrict__ WhhB,  // [kH][kH]   = W_hh (already B^T layout)
    const bf16_t* __restrict__ W1T,   // [kH][kH]   = W1 transposed
    const bf16_t* __restrict__ W2T,   // [kOUT][kH] = W2 transposed
    float* __restrict__ out) {
  extern __shared__ char smem_raw[];
  SmcLds& L = *reinterpret_cast<SmcLds*>(smem_raw);
  const int b   = blockIdx.x;
  const int tid = threadIdx.x;

  // ---- preload per-block constants ----
  for (int i = tid; i < kH; i += 256) {
    L.sbih[i] = b_ih[i] + b_hh[i];
    L.sb1[i]  = b1[i];
    L.ssx[i]  = __builtin_amdgcn_sqrtf(sigx[i]);
  }
  if (tid < kOUT) {
    L.sb2[tid]  = b2[tid];
    L.sisy[tid] = __builtin_amdgcn_rcpf(sigy[tid]);
  }
  // ---- x0 ~ N(0,1) ----
  for (int i = tid; i < kNP * kH; i += 256)
    L.xbuf[i] = (bf16_t)gaussian(0x51ED270Bu + (uint32_t)(b * kNP * kH + i));
  __syncthreads();

  for (int t = 0; t < kT; ++t) {
    // prefetch next step's u/y rows (global_prefetch_b8)
    {
      const int tnx = (t + 1 < kT) ? t + 1 : t;
      if (tid < 16)
        __builtin_prefetch(&u[(size_t)(tnx * kBS + b) * kH + tid * 16], 0, 0);
      else if (tid < 20)
        __builtin_prefetch(&y[(size_t)(tnx * kBS + b) * kOUT + (tid - 16) * 16], 0, 0);
    }
    // u_t / y_t rows
    for (int i = tid; i < kH; i += 256) L.su[i] = u[(size_t)(t * kBS + b) * kH + i];
    if (tid < kOUT) L.syrow[tid] = y[(size_t)(t * kBS + b) * kOUT + tid];
    __syncthreads();

    // u-projection: uproj[h] = sum_i u[i]*W_ih[h,i] + b_ih[h] + b_hh[h]
    for (int hh = tid; hh < kH; hh += 256) {
      const float* wr = W_ih + hh * kH;
      float s = L.sbih[hh];
#pragma unroll 8
      for (int i = 0; i < kH; ++i) s += L.su[i] * wr[i];
      L.suproj[hh] = s;
    }
    __syncthreads();

    // GEMM1: h = tanh(x @ W_hh^T + uproj)
    gemm_block<0, kH>(L.xbuf, WhhB, L.hbuf, tid, L.suproj, nullptr);
    __syncthreads();

    // noise pass: h += N(0,1) * sqrt(sigma_x)   (rolled loop; one static copy
    // of the Box-Muller code instead of 64 inlined into the GEMM epilogue)
    {
      const uint32_t nb = 0xA5130000u +
          (uint32_t)(t * kBS + b) * (uint32_t)(kNP * kH);
      for (int i = tid; i < kNP * kH; i += 256) {
        const int col = i & (kH - 1);
        float v = (float)L.hbuf[i] + gaussian(nb + (uint32_t)i) * L.ssx[col];
        L.hbuf[i] = (bf16_t)v;
      }
    }
    __syncthreads();

    // GEMM2: z = relu(h @ W1 + b1)   (z overwrites xbuf; x dead after GEMM1)
    gemm_block<1, kH>(L.hbuf, W1T, L.xbuf, tid, L.sb1, nullptr);
    __syncthreads();

    // GEMM3: y_hat = z @ W2 + b2 -> ybuf + global out
    float* gout = out + (size_t)(t * kBS + b) * (kNP * kOUT);
    gemm_block<2, kOUT>(L.xbuf, W2T, L.ybuf, tid, L.sb2, gout);
    __syncthreads();

    // per-particle log-prob
    if (tid < kNP) {
      const float* yr = &L.ybuf[tid * kOUT];
      float s = 0.f;
#pragma unroll 8
      for (int d = 0; d < kOUT; ++d) {
        float df = yr[d] - L.syrow[d];
        s += df * df * L.sisy[d];
      }
      L.slp[tid] = -0.5f * s;
    }
    __syncthreads();

    // softmax cdf over particles (64 values; single-lane scan is cheap)
    if (tid == 0) {
      float m = L.slp[0];
      for (int i = 1; i < kNP; ++i) m = fmaxf(m, L.slp[i]);
      float acc = 0.f;
      for (int i = 0; i < kNP; ++i) { acc += __expf(L.slp[i] - m); L.scdf[i] = acc; }
      L.stot = acc;
    }
    __syncthreads();

    // categorical resampling indices
    if (tid < kNP) {
      float r = uniform01(hash_u32(0xC0FFEE11u ^
                 (uint32_t)((t * kBS + b) * kNP + tid))) * L.stot;
      int idx = kNP - 1;
      for (int j = 0; j < kNP; ++j) {
        if (L.scdf[j] >= r) { idx = j; break; }
      }
      L.sI[tid] = idx;
    }
    __syncthreads();

    // gather: x_new[n] = h[I[n]]  (bf16 rows, 16B vector copies, no aliasing)
    {
      constexpr int VPR = kH / 8;  // uint4 (8 bf16) chunks per row
      const uint4* src = (const uint4*)L.hbuf;
      uint4* dst = (uint4*)L.xbuf;
      for (int i = tid; i < kNP * VPR; i += 256) {
        int n = i / VPR, c = i % VPR;
        dst[n * VPR + c] = src[L.sI[n] * VPR + c];
      }
    }
    __syncthreads();
  }
}

// ---------------- one-time weight prep: f32 -> bf16 (+transpose) ----------------
__global__ void prep_weights(const float* __restrict__ Whh,
                             const float* __restrict__ W1,
                             const float* __restrict__ W2,
                             bf16_t* __restrict__ WhhB,
                             bf16_t* __restrict__ W1T,
                             bf16_t* __restrict__ W2T) {
  int idx = blockIdx.x * blockDim.x + threadIdx.x;
  if (idx < kH * kH) {
    WhhB[idx] = (bf16_t)Whh[idx];               // W_hh rows are already B^T layout
    int n = idx / kH, k = idx % kH;
    W1T[idx] = (bf16_t)W1[k * kH + n];          // W1T[n][k] = W1[k][n]
  }
  if (idx < kOUT * kH) {
    int o = idx / kH, k = idx % kH;
    W2T[idx] = (bf16_t)W2[k * kOUT + o];        // W2T[o][k] = W2[k][o]
  }
}

extern "C" void kernel_launch(void* const* d_in, const int* in_sizes, int n_in,
                              void* d_out, int out_size, void* d_ws, size_t ws_size,
                              hipStream_t stream) {
  const float* u    = (const float*)d_in[0];
  const float* y    = (const float*)d_in[1];
  const float* W_ih = (const float*)d_in[2];
  const float* W_hh = (const float*)d_in[3];
  const float* b_ih = (const float*)d_in[4];
  const float* b_hh = (const float*)d_in[5];
  const float* W1   = (const float*)d_in[6];
  const float* b1   = (const float*)d_in[7];
  const float* W2   = (const float*)d_in[8];
  const float* b2   = (const float*)d_in[9];
  const float* sigx = (const float*)d_in[10];
  const float* sigy = (const float*)d_in[11];
  float* out = (float*)d_out;

  // Workspace: bf16 weight copies (288KB total).
  bf16_t* WhhB = (bf16_t*)d_ws;                  // kH*kH
  bf16_t* W1T  = WhhB + (size_t)kH * kH;         // kH*kH
  bf16_t* W2T  = W1T + (size_t)kH * kH;          // kOUT*kH

  prep_weights<<<dim3((kH * kH + 255) / 256), dim3(256), 0, stream>>>(
      W_hh, W1, W2, WhhB, W1T, W2T);

  smcn_kernel<<<dim3(kBS), dim3(256), sizeof(SmcLds), stream>>>(
      u, y, W_ih, b_ih, b_hh, b1, b2, sigx, sigy, WhhB, W1T, W2T, out);
}
